// RzLinear_54322746360525
// MI455X (gfx1250) — matrix operand order repaired and checked by reading, not compiled, not run on hardware
//
#include <hip/hip_runtime.h>
#include <stdint.h>

// ---------------- problem constants (match reference) ----------------
#define TOKENS  8192
#define IN_DIM  4096
#define OUT_DIM 4096
#define CHUNKSZ 32
#define MEM_SZ  262144
#define HMOD    (MEM_SZ - CHUNKSZ)      // 262112, compile-time -> magic-number mod
#define PRIME_C 2038074743ull           // random_numbers[0], fixed by reference

// ---------------- tiling ----------------
#define MT 256          // C tile rows per workgroup
#define NT 128          // C tile cols per workgroup
#define KT 32           // K step (== CHUNK == bf16 WMMA K)
#define KP 40           // LDS row stride in bf16 elems (80 B: 16B-aligned, padded)

typedef __attribute__((ext_vector_type(16))) __bf16 v16bf;
typedef __attribute__((ext_vector_type(8)))  float  v8f;

__device__ __forceinline__ void split_bf16(float f, ushort& hb, ushort& lb) {
    uint32_t u = __float_as_uint(f);
    uint32_t h = (u + 0x7FFFu + ((u >> 16) & 1u)) >> 16;   // RNE round to bf16
    float fh   = __uint_as_float(h << 16);
    lb = (ushort)(__float_as_uint(f - fh) >> 16);          // residual, truncated
    hb = (ushort)h;
}

// A fragment (16-bit 16x32): lane m = lane&15, half = lane>>4
__device__ __forceinline__ v16bf load_frag_a(const ushort* row, int kh) {
    const uint4* p = (const uint4*)(row + kh * 8);
    union { uint4 q[2]; v16bf v; } u;
    u.q[0] = p[0];       // K = kh*8 .. +7
    u.q[1] = p[2];       // K = 16+kh*8 .. +7  (+32 bytes)
    return u.v;
}

// B fragment (16-bit 32x16): lane n = lane&15; K = kh*8..+7 and 16+kh*8..+7
__device__ __forceinline__ v16bf load_frag_b(const ushort* row, int kh) {
    const uint4* p = (const uint4*)(row + kh * 8);
    union { uint4 q[2]; v16bf v; } u;
    u.q[0] = p[0];
    u.q[1] = p[2];
    return u.v;
}

__device__ __forceinline__ v8f wmma_bf16(v16bf a, v16bf b, v8f c) {
    return __builtin_amdgcn_wmma_f32_16x16x32_bf16(false, a, false, b,
                                                   (short)0, c, false, false);
}

// Registers staging one k-tile (globals landed before the WMMA block)
struct Stage {
    float4 a[8];    // 8 x float4: this thread's slice of the 256x32 A tile
    float  w[16];   // 4 chunk-parts x 4 floats of the gathered W tile
};

__device__ __forceinline__ void stage_load(Stage& st,
        const float* __restrict__ A, const float* __restrict__ HW,
        int m0, int n0, int k0, int t, int rowp, int kq,
        unsigned long long a_mul, unsigned long long b_add) {
#pragma unroll
    for (int p = 0; p < 8; ++p) {
        const int m = (p << 5) + rowp;
        const float* gp = A + (size_t)(m0 + m) * IN_DIM + (k0 + kq);
        st.a[p] = *(const float4*)gp;
        __builtin_prefetch(gp + KT, 0, 0);   // pull tile k0+KT toward caches
    }
#pragma unroll
    for (int q = 0; q < 4; ++q) {
        const int s     = t + (q << 8);     // 0..1023
        const int chunk = s >> 3;           // 0..127 (32 k-rows x 4 n-chunks)
        const int part  = s & 7;
        const int krow  = chunk >> 2;
        const int jj    = chunk & 3;
        const unsigned long long c =
            (unsigned long long)((k0 + krow) * 128 + (n0 >> 5) + jj);
        const unsigned long long hh = (c * a_mul + b_add) % PRIME_C;
        const unsigned h = (unsigned)(hh % (unsigned long long)HMOD);
        const float* src = HW + h + (part << 2);
#pragma unroll
        for (int i = 0; i < 4; ++i) st.w[(q << 2) + i] = src[i];
    }
}

__device__ __forceinline__ void stage_store(const Stage& st,
        ushort (*aHi)[KP], ushort (*aLo)[KP],
        ushort (*wHi)[KP], ushort (*wLo)[KP],
        int t, int rowp, int kq) {
#pragma unroll
    for (int p = 0; p < 8; ++p) {
        const int m = (p << 5) + rowp;
        const float xs[4] = {st.a[p].x, st.a[p].y, st.a[p].z, st.a[p].w};
#pragma unroll
        for (int i = 0; i < 4; ++i) {
            ushort hb, lb;
            split_bf16(xs[i], hb, lb);
            aHi[m][kq + i] = hb;
            aLo[m][kq + i] = lb;
        }
    }
#pragma unroll
    for (int q = 0; q < 4; ++q) {
        const int s     = t + (q << 8);
        const int chunk = s >> 3;
        const int part  = s & 7;
        const int krow  = chunk >> 2;
        const int jj    = chunk & 3;
        const int nn    = (jj << 5) + (part << 2);   // local n, tile stored [n][k]
#pragma unroll
        for (int i = 0; i < 4; ++i) {
            ushort hb, lb;
            split_bf16(st.w[(q << 2) + i], hb, lb);
            wHi[nn + i][krow] = hb;
            wLo[nn + i][krow] = lb;
        }
    }
}

// HIP semantics: (MAX_THREADS_PER_BLOCK, MIN_WARPS_PER_EXECUTION_UNIT).
// 4 waves/SIMD -> 256-VGPR budget -> 2 WGs/WGP (LDS 2x120KB <= 320KB).
__global__ __launch_bounds__(256, 4)
void rz_gemm_bf16x3(const float* __restrict__ A,
                    const float* __restrict__ HW,
                    const float* __restrict__ bias,
                    const long long* __restrict__ rn,
                    float* __restrict__ C) {
    // double-buffered hi/lo bf16 staging (120 KB of the 320 KB WGP LDS)
    __shared__ ushort sAhi[2][MT][KP];
    __shared__ ushort sAlo[2][MT][KP];
    __shared__ ushort sWhi[2][NT][KP];
    __shared__ ushort sWlo[2][NT][KP];

    const int t    = threadIdx.x;
    const int lane = t & 31;
    const int wave = t >> 5;
    const int wm   = wave & 3;        // 4 waves along M, 64 rows each
    const int wn   = wave >> 2;       // 2 waves along N, 64 cols each
    const int kh   = lane >> 4;
    const int l16  = lane & 15;

    const int m0 = blockIdx.y * MT;
    const int n0 = blockIdx.x * NT;

    const unsigned long long a_mul = (unsigned long long)rn[1];
    const unsigned long long b_add = (unsigned long long)rn[2];

    const v8f vzero = {0.f,0.f,0.f,0.f,0.f,0.f,0.f,0.f};
    v8f acc[4][4];
#pragma unroll
    for (int i = 0; i < 4; ++i)
#pragma unroll
        for (int j = 0; j < 4; ++j) acc[i][j] = vzero;

    const int rowp = t >> 3;          // 0..31
    const int kq   = (t & 7) << 2;    // 0,4,..,28

    // ---- prologue: stage k0 = 0 into buffer 0 ----
    Stage st;
    stage_load(st, A, HW, m0, n0, 0, t, rowp, kq, a_mul, b_add);
    stage_store(st, sAhi[0], sAlo[0], sWhi[0], sWlo[0], t, rowp, kq);
    __syncthreads();

    for (int k0 = 0; k0 < IN_DIM; k0 += KT) {
        const int  cur  = (k0 >> 5) & 1;
        const int  nxt  = cur ^ 1;
        const bool more = (k0 + KT) < IN_DIM;

        // issue next tile's global loads; latency hidden under the WMMA block
        if (more)
            stage_load(st, A, HW, m0, n0, k0 + KT, t, rowp, kq, a_mul, b_add);

        // ---- pass 1: Ahi terms (32 WMMAs): acc += Ahi*Whi + Ahi*Wlo ----
        v16bf af[4];
#pragma unroll
        for (int i = 0; i < 4; ++i) {
            const int mrow = wm * 64 + i * 16 + l16;
            af[i] = load_frag_a(&sAhi[cur][mrow][0], kh);
        }
#pragma unroll
        for (int j = 0; j < 4; ++j) {
            const int nrow = wn * 64 + j * 16 + l16;
            const v16bf wh = load_frag_b(&sWhi[cur][nrow][0], kh);
            const v16bf wl = load_frag_b(&sWlo[cur][nrow][0], kh);
#pragma unroll
            for (int i = 0; i < 4; ++i) {
                acc[i][j] = wmma_bf16(af[i], wh, acc[i][j]);  // hi*hi
                acc[i][j] = wmma_bf16(af[i], wl, acc[i][j]);  // hi*lo
            }
        }

        // ---- pass 2: Alo terms (16 WMMAs): acc += Alo*Whi ----
#pragma unroll
        for (int i = 0; i < 4; ++i) {
            const int mrow = wm * 64 + i * 16 + l16;
            af[i] = load_frag_a(&sAlo[cur][mrow][0], kh);   // reuse registers
        }
#pragma unroll
        for (int j = 0; j < 4; ++j) {
            const int nrow = wn * 64 + j * 16 + l16;
            const v16bf wh = load_frag_b(&sWhi[cur][nrow][0], kh);
#pragma unroll
            for (int i = 0; i < 4; ++i) {
                acc[i][j] = wmma_bf16(af[i], wh, acc[i][j]);  // lo*hi
            }
        }

        // split + store next tile into the alternate buffer
        if (more)
            stage_store(st, sAhi[nxt], sAlo[nxt], sWhi[nxt], sWlo[nxt],
                        t, rowp, kq);

        __syncthreads();   // one barrier per k-step (ping-pong)
    }

    // ---- epilogue: bias + store (C/D layout: M = vgpr + 8*half, N = lane&15) ----
    float bv[4];
#pragma unroll
    for (int j = 0; j < 4; ++j) bv[j] = bias[n0 + wn * 64 + j * 16 + l16];

#pragma unroll
    for (int i = 0; i < 4; ++i) {
        const int mbase = m0 + wm * 64 + i * 16 + kh * 8;
#pragma unroll
        for (int j = 0; j < 4; ++j) {
            const int col = n0 + wn * 64 + j * 16 + l16;
#pragma unroll
            for (int r = 0; r < 8; ++r) {
                C[(size_t)(mbase + r) * OUT_DIM + col] = acc[i][j][r] + bv[j];
            }
        }
    }
}

extern "C" void kernel_launch(void* const* d_in, const int* in_sizes, int n_in,
                              void* d_out, int out_size, void* d_ws, size_t ws_size,
                              hipStream_t stream) {
    (void)in_sizes; (void)n_in; (void)d_ws; (void)ws_size; (void)out_size;
    const float*     A    = (const float*)d_in[0];
    const float*     HW   = (const float*)d_in[1];
    const float*     bias = (const float*)d_in[2];
    const long long* rn   = (const long long*)d_in[3];   // int64 per reference
    float*           C    = (float*)d_out;

    dim3 grid(OUT_DIM / NT, TOKENS / MT);   // (32, 32) = 1024 workgroups
    rz_gemm_bf16x3<<<grid, dim3(256), 0, stream>>>(A, HW, bias, rn, C);
}
